// BottleneckAttention3D_22789096473254
// MI455X (gfx1250) — matrix operand order repaired and checked by reading, not compiled, not run on hardware
//
#include <hip/hip_runtime.h>
#include <hip/hip_bf16.h>
#include <stdint.h>

typedef __attribute__((ext_vector_type(16))) __bf16 v16bf;
typedef __attribute__((ext_vector_type(8)))  float  v8f;

#define C_DIM   128
#define N_DIM   4096
#define B_DIM   2
#define GROUPS  8
#define CPG     16                 // channels per group
#define GN_ELEMS (CPG * N_DIM)     // 65536 contiguous floats per (b,g)
#define TILE    128
#define NTILES  (N_DIM / TILE)     // 32
#define SPLIT   4                  // key-dimension splits for flash attention
#define KT_PER_SPLIT (NTILES / SPLIT)   // 8
#define ATTN_SCALE 0.08838834764831845f // 1/sqrt(128)
#define GN_EPS  1e-5f

// ---------- helpers ----------

static __device__ __forceinline__ __bf16 f2bf(float f) {
  // round-to-nearest-even f32 -> bf16
  unsigned u = __float_as_uint(f);
  unsigned r = (u + 0x7fffu + ((u >> 16) & 1u)) >> 16;
  unsigned short s = (unsigned short)r;
  __bf16 b;
  __builtin_memcpy(&b, &s, 2);
  return b;
}

// ---- CDNA5 async global->LDS path (ASYNCcnt, bypasses VGPRs) ----
static __device__ __forceinline__ void async_ld16(void* lds_dst, const void* gsrc) {
  unsigned l = (unsigned)(uintptr_t)lds_dst;   // generic LDS ptr low 32 bits = LDS offset
  asm volatile("global_load_async_to_lds_b128 %0, %1, off"
               :: "v"(l), "v"(gsrc) : "memory");
}
static __device__ __forceinline__ void wait_async() {
  asm volatile("s_wait_asynccnt 0" ::: "memory");
}

// block-cooperative async 16B-granule copy (both 16B aligned, nelem % 8 == 0)
static __device__ __forceinline__ void cp16_async(__bf16* dst, const __bf16* src, int nelem) {
  uint4* d = (uint4*)dst;
  const uint4* s = (const uint4*)src;
  int n = nelem >> 3;
  for (int i = threadIdx.x; i < n; i += blockDim.x) async_ld16(d + i, s + i);
  wait_async();
}

union BF16x16 { uint4 u[2]; v16bf v; };

// A fragment, 16x32 bf16, source row-major [M][K] with leading dim ld.
static __device__ __forceinline__ v16bf load_a_frag(const __bf16* base, int row0, int ld, int k) {
  int lane = threadIdx.x & 31;
  int half = lane >> 4, l16 = lane & 15;
  const __bf16* p = base + (row0 + l16) * ld + k + half * 8;
  BF16x16 r;
  r.u[0] = *(const uint4*)p;
  r.u[1] = *(const uint4*)(p + 16);
  return r.v;
}

// B fragment, 32x16 bf16, source stored as B^T row-major [N'][K] with leading dim ld.
static __device__ __forceinline__ v16bf load_b_frag(const __bf16* base, int col0, int ld, int k) {
  int lane = threadIdx.x & 31;
  int half = lane >> 4, l16 = lane & 15;
  const __bf16* p = base + (col0 + l16) * ld + k + half * 8;
  BF16x16 r;
  r.u[0] = *(const uint4*)p;
  r.u[1] = *(const uint4*)(p + 16);
  return r.v;
}

static __device__ __forceinline__ v8f wmma_bf16(v16bf a, v16bf b, v8f c) {
  return __builtin_amdgcn_wmma_f32_16x16x32_bf16(false, a, false, b, (short)0, c, false, false);
}

// D/C fragment mapping: element j of lane L -> row (j + (L/16)*8), col (L%16)

// ---------- kernel 0: weights -> bf16 ----------
__global__ void k_convert_w(const float* wq, const float* wk, const float* wv, const float* wp,
                            __bf16* wqb, __bf16* wkb, __bf16* wvb, __bf16* wpb) {
  int i = blockIdx.x * blockDim.x + threadIdx.x;
  if (i < C_DIM * C_DIM) {
    wqb[i] = f2bf(wq[i]);
    wkb[i] = f2bf(wk[i]);
    wvb[i] = f2bf(wv[i]);
    wpb[i] = f2bf(wp[i]);
  }
}

// ---------- kernel 1: groupnorm statistics ----------
__global__ void k_gn_stats(const float* __restrict__ x, float2* __restrict__ stats) {
  int bg = blockIdx.x;
  const float* xp = x + (size_t)bg * GN_ELEMS;   // [b][g] slab is contiguous
  float s = 0.f, ss = 0.f;
  for (int i = threadIdx.x; i < GN_ELEMS; i += blockDim.x) {
    float v = xp[i];
    s += v;
    ss += v * v;
  }
  __shared__ float rs[256], rss[256];
  rs[threadIdx.x] = s;
  rss[threadIdx.x] = ss;
  __syncthreads();
  for (int o = 128; o > 0; o >>= 1) {
    if ((int)threadIdx.x < o) {
      rs[threadIdx.x]  += rs[threadIdx.x + o];
      rss[threadIdx.x] += rss[threadIdx.x + o];
    }
    __syncthreads();
  }
  if (threadIdx.x == 0) {
    float mean = rs[0] * (1.0f / GN_ELEMS);
    float var  = rss[0] * (1.0f / GN_ELEMS) - mean * mean;
    stats[bg] = make_float2(mean, rsqrtf(var + GN_EPS));
  }
}

// ---------- kernel 2: groupnorm apply -> h^T [B][N][C] bf16 ----------
__global__ void k_gn_apply(const float* __restrict__ x, const float* __restrict__ gamma,
                           const float* __restrict__ beta, const float2* __restrict__ stats,
                           __bf16* __restrict__ hT) {
  int i = blockIdx.x * blockDim.x + threadIdx.x;   // flat over B*C*N
  if (i >= B_DIM * C_DIM * N_DIM) return;
  int n = i % N_DIM;
  int c = (i / N_DIM) % C_DIM;
  int b = i / (N_DIM * C_DIM);
  float2 st = stats[b * GROUPS + c / CPG];
  float v = (x[i] - st.x) * st.y * gamma[c] + beta[c];
  hT[((size_t)b * N_DIM + n) * C_DIM + c] = f2bf(v);
}

// ---------- kernel 3: QKV projections (WMMA) ----------
// grid: (NTILES, 3, B). Stores: Q,K as [N][C] (Q pre-scaled); V as [C][N].
__global__ void __launch_bounds__(256) k_qkv(const __bf16* __restrict__ hT,
                                             const __bf16* __restrict__ wqb,
                                             const __bf16* __restrict__ wkb,
                                             const __bf16* __restrict__ wvb,
                                             const float* __restrict__ bq,
                                             const float* __restrict__ bk,
                                             const float* __restrict__ bv,
                                             __bf16* __restrict__ Q,
                                             __bf16* __restrict__ K,
                                             __bf16* __restrict__ V) {
  __shared__ __bf16 sW[TILE * C_DIM];
  __shared__ __bf16 sH[TILE * C_DIM];
  int n0 = blockIdx.x * TILE;
  int which = blockIdx.y;  // 0=q 1=k 2=v
  int b = blockIdx.z;
  const __bf16* w = (which == 0) ? wqb : (which == 1) ? wkb : wvb;
  const float* bias = (which == 0) ? bq : (which == 1) ? bk : bv;
  cp16_async(sW, w, TILE * C_DIM);
  cp16_async(sH, hT + ((size_t)b * N_DIM + n0) * C_DIM, TILE * C_DIM);
  __syncthreads();

  int wid = threadIdx.x >> 5;
  int lane = threadIdx.x & 31, half = lane >> 4, l16 = lane & 15;
  v8f zero = {0.f, 0.f, 0.f, 0.f, 0.f, 0.f, 0.f, 0.f};
  v8f acc[8];
  for (int t = 0; t < 8; ++t) acc[t] = zero;

  for (int ks = 0; ks < 4; ++ks) {
    v16bf a = load_a_frag(sW, wid * 16, C_DIM, ks * 32);
    for (int nt = 0; nt < 8; ++nt) {
      v16bf bf = load_b_frag(sH, nt * 16, C_DIM, ks * 32);
      acc[nt] = wmma_bf16(a, bf, acc[nt]);
    }
  }

  for (int nt = 0; nt < 8; ++nt) {
    for (int j = 0; j < 8; ++j) {
      int m  = wid * 16 + half * 8 + j;   // output channel
      int nl = nt * 16 + l16;             // spatial within tile
      float val = acc[nt][j] + bias[m];
      if (which == 0) {
        Q[((size_t)b * N_DIM + (n0 + nl)) * C_DIM + m] = f2bf(val * ATTN_SCALE);
      } else if (which == 1) {
        K[((size_t)b * N_DIM + (n0 + nl)) * C_DIM + m] = f2bf(val);
      } else {
        V[((size_t)b * C_DIM + m) * N_DIM + (n0 + nl)] = f2bf(val);
      }
    }
  }
}

// ---------- kernel 4: split-K flash attention (WMMA + online softmax) ----------
// grid: (NTILES, SPLIT, B). Each block: 128 query rows x 8 key tiles.
// Emits unnormalized partials: Opart fp32 [S][B][N][C], ml float2 [S][B][N].
__global__ void __launch_bounds__(256) k_attn(const __bf16* __restrict__ Q,
                                              const __bf16* __restrict__ K,
                                              const __bf16* __restrict__ V,
                                              float* __restrict__ Opart,
                                              float2* __restrict__ ml) {
  __shared__ __bf16 sQ[TILE * C_DIM];
  __shared__ __bf16 sKV[TILE * C_DIM];   // K tile, then reused for V tile
  __shared__ __bf16 sP[TILE * TILE];
  int q0 = blockIdx.x * TILE;
  int split = blockIdx.y;
  int b = blockIdx.z;
  cp16_async(sQ, Q + ((size_t)b * N_DIM + q0) * C_DIM, TILE * C_DIM);

  int wid = threadIdx.x >> 5;
  int lane = threadIdx.x & 31, half = lane >> 4, l16 = lane & 15;
  v8f zero = {0.f, 0.f, 0.f, 0.f, 0.f, 0.f, 0.f, 0.f};
  v8f accO[8];
  float rowmax[8], rowsum[8];
  for (int t = 0; t < 8; ++t) { accO[t] = zero; rowmax[t] = -1e30f; rowsum[t] = 0.f; }

  for (int kt = split * KT_PER_SPLIT; kt < (split + 1) * KT_PER_SPLIT; ++kt) {
    int j0 = kt * TILE;
    __syncthreads();                                   // prior PV done with sKV/sP
    cp16_async(sKV, K + ((size_t)b * N_DIM + j0) * C_DIM, TILE * C_DIM);
    __syncthreads();

    // S = Q * K^T (1/sqrt(C) folded into Q)
    v8f accS[8];
    for (int t = 0; t < 8; ++t) accS[t] = zero;
    for (int ks = 0; ks < 4; ++ks) {
      v16bf a = load_a_frag(sQ, wid * 16, C_DIM, ks * 32);
      for (int nt = 0; nt < 8; ++nt) {
        v16bf bf = load_b_frag(sKV, nt * 16, C_DIM, ks * 32);
        accS[nt] = wmma_bf16(a, bf, accS[nt]);
      }
    }

    // per-row online softmax; row (j,half) is shared by a 16-lane half-wave
    for (int j = 0; j < 8; ++j) {
      float sv[8];
      float tmax = -1e30f;
      for (int nt = 0; nt < 8; ++nt) { sv[nt] = accS[nt][j]; tmax = fmaxf(tmax, sv[nt]); }
      for (int off = 8; off >= 1; off >>= 1)
        tmax = fmaxf(tmax, __shfl_xor(tmax, off, 32));
      float mnew = fmaxf(rowmax[j], tmax);
      float corr = __expf(rowmax[j] - mnew);
      float psum = 0.f;
      int prow = wid * 16 + half * 8 + j;
      for (int nt = 0; nt < 8; ++nt) {
        float p = __expf(sv[nt] - mnew);
        psum += p;
        sP[prow * TILE + nt * 16 + l16] = f2bf(p);
      }
      for (int off = 8; off >= 1; off >>= 1)
        psum += __shfl_xor(psum, off, 32);
      rowsum[j] = rowsum[j] * corr + psum;
      rowmax[j] = mnew;
      for (int nt = 0; nt < 8; ++nt) accO[nt][j] *= corr;
    }
    __syncthreads();   // sP complete, S-phase done reading sKV

    // load V tile [c][m_local] (rows strided by N in global)
    for (int idx = threadIdx.x; idx < (TILE * C_DIM) / 8; idx += blockDim.x) {
      int c = idx >> 4;       // 16 uint4 per 128-element row
      int u = idx & 15;
      async_ld16(((uint4*)sKV) + idx,
                 ((const uint4*)(V + ((size_t)b * C_DIM + c) * N_DIM + j0)) + u);
    }
    wait_async();
    __syncthreads();

    // O += P * V  (A = P rows [n][m], B^T = V [c][m])
    for (int ks = 0; ks < 4; ++ks) {
      v16bf a = load_a_frag(sP, wid * 16, TILE, ks * 32);
      for (int nt = 0; nt < 8; ++nt) {
        v16bf bf = load_b_frag(sKV, nt * 16, TILE, ks * 32);
        accO[nt] = wmma_bf16(a, bf, accO[nt]);
      }
    }
  }

  // store unnormalized split partials
  for (int j = 0; j < 8; ++j) {
    int row = q0 + wid * 16 + half * 8 + j;
    size_t rbase = ((size_t)split * B_DIM + b) * N_DIM + row;
    for (int nt = 0; nt < 8; ++nt) {
      Opart[rbase * C_DIM + nt * 16 + l16] = accO[nt][j];
    }
    if (l16 == 0) ml[rbase] = make_float2(rowmax[j], rowsum[j]);
  }
}

// ---------- kernel 4b: combine split partials -> O2 [N][C] bf16 ----------
// grid: B*N blocks of C threads
__global__ void k_combine(const float* __restrict__ Opart, const float2* __restrict__ ml,
                          __bf16* __restrict__ O2) {
  int bn = blockIdx.x;     // b*N + n
  int c = threadIdx.x;
  float m[SPLIT], l[SPLIT];
  float M = -1e30f;
  for (int s = 0; s < SPLIT; ++s) {
    float2 v = ml[(size_t)s * B_DIM * N_DIM + bn];
    m[s] = v.x;
    l[s] = v.y;
    M = fmaxf(M, m[s]);
  }
  float L = 0.f, o = 0.f;
  for (int s = 0; s < SPLIT; ++s) {
    float w = __expf(m[s] - M);
    L += l[s] * w;
    o += Opart[((size_t)s * B_DIM * N_DIM + bn) * C_DIM + c] * w;
  }
  O2[(size_t)bn * C_DIM + c] = f2bf(o / L);
}

// ---------- kernel 5: output projection + residual ----------
__global__ void __launch_bounds__(256) k_proj(const __bf16* __restrict__ O2,
                                              const __bf16* __restrict__ wpb,
                                              const float* __restrict__ bp,
                                              const float* __restrict__ x,
                                              float* __restrict__ out) {
  __shared__ __bf16 sW[TILE * C_DIM];
  __shared__ __bf16 sO[TILE * C_DIM];
  int n0 = blockIdx.x * TILE;
  int b = blockIdx.y;
  cp16_async(sW, wpb, TILE * C_DIM);
  cp16_async(sO, O2 + ((size_t)b * N_DIM + n0) * C_DIM, TILE * C_DIM);
  __syncthreads();

  int wid = threadIdx.x >> 5;
  int lane = threadIdx.x & 31, half = lane >> 4, l16 = lane & 15;
  v8f zero = {0.f, 0.f, 0.f, 0.f, 0.f, 0.f, 0.f, 0.f};
  v8f acc[8];
  for (int t = 0; t < 8; ++t) acc[t] = zero;

  for (int ks = 0; ks < 4; ++ks) {
    v16bf a = load_a_frag(sW, wid * 16, C_DIM, ks * 32);
    for (int nt = 0; nt < 8; ++nt) {
      v16bf bf = load_b_frag(sO, nt * 16, C_DIM, ks * 32);
      acc[nt] = wmma_bf16(a, bf, acc[nt]);
    }
  }

  for (int nt = 0; nt < 8; ++nt) {
    for (int j = 0; j < 8; ++j) {
      int m  = wid * 16 + half * 8 + j;
      int nl = nt * 16 + l16;
      size_t idx = ((size_t)b * C_DIM + m) * N_DIM + (n0 + nl);
      out[idx] = x[idx] + acc[nt][j] + bp[m];
    }
  }
}

// ---------- host launch ----------
extern "C" void kernel_launch(void* const* d_in, const int* in_sizes, int n_in,
                              void* d_out, int out_size, void* d_ws, size_t ws_size,
                              hipStream_t stream) {
  (void)in_sizes; (void)n_in; (void)out_size; (void)ws_size;
  const float* x     = (const float*)d_in[0];
  const float* gamma = (const float*)d_in[1];
  const float* beta  = (const float*)d_in[2];
  const float* wq    = (const float*)d_in[3];
  const float* bq    = (const float*)d_in[4];
  const float* wk    = (const float*)d_in[5];
  const float* bk    = (const float*)d_in[6];
  const float* wv    = (const float*)d_in[7];
  const float* bv    = (const float*)d_in[8];
  const float* wp    = (const float*)d_in[9];
  const float* bp    = (const float*)d_in[10];
  float* out = (float*)d_out;

  char* ws = (char*)d_ws;
  const size_t WBYTES = (size_t)C_DIM * C_DIM * 2;         // 32 KB each
  const size_t MBYTES = (size_t)B_DIM * N_DIM * C_DIM * 2; // 2 MB each bf16 tensor
  __bf16* wqb  = (__bf16*)(ws);
  __bf16* wkb  = (__bf16*)(ws + WBYTES);
  __bf16* wvb  = (__bf16*)(ws + 2 * WBYTES);
  __bf16* wpb  = (__bf16*)(ws + 3 * WBYTES);
  float2* stats = (float2*)(ws + 4 * WBYTES);              // 128 B
  size_t base = 262144;                                    // 256 KB aligned region
  __bf16* hT = (__bf16*)(ws + base);
  __bf16* Qs = (__bf16*)(ws + base + MBYTES);
  __bf16* Ks = (__bf16*)(ws + base + 2 * MBYTES);
  __bf16* Vs = (__bf16*)(ws + base + 3 * MBYTES);
  __bf16* O2 = (__bf16*)(ws + base + 4 * MBYTES);
  float2* ml  = (float2*)(ws + base + 5 * MBYTES);                   // 256 KB
  float*  Opart = (float*)(ws + base + 5 * MBYTES + 262144);         // 16 MB

  k_convert_w<<<64, 256, 0, stream>>>(wq, wk, wv, wp, wqb, wkb, wvb, wpb);
  k_gn_stats<<<B_DIM * GROUPS, 256, 0, stream>>>(x, stats);
  k_gn_apply<<<(B_DIM * C_DIM * N_DIM) / 256, 256, 0, stream>>>(x, gamma, beta, stats, hT);

  dim3 gqkv(NTILES, 3, B_DIM);
  k_qkv<<<gqkv, 256, 0, stream>>>(hT, wqb, wkb, wvb, bq, bk, bv, Qs, Ks, Vs);

  dim3 gat(NTILES, SPLIT, B_DIM);
  k_attn<<<gat, 256, 0, stream>>>(Qs, Ks, Vs, Opart, ml);
  k_combine<<<B_DIM * N_DIM, C_DIM, 0, stream>>>(Opart, ml, O2);

  dim3 gproj(NTILES, B_DIM);
  k_proj<<<gproj, 256, 0, stream>>>(O2, wpb, bp, x, out);
}
